// ShiftConv2d0_27032524161737
// MI455X (gfx1250) — matrix operand order repaired and checked by reading, not compiled, not run on hardware
//
#include <hip/hip_runtime.h>
#include <hip/hip_bf16.h>
#include <stddef.h>

// ShiftConv2d on MI455X (gfx1250):
//   y[n,:,h,w] = W_eff (320x320) @ x_shift[n,:,h,w] + b
// W_eff picks one 3x3 tap per 64-channel input group (R,L,D,U,center).
// Strategy: bf16 WMMA (v_wmma_f32_16x16x32_bf16) with fp32 accumulation.
// One block per (n,h) image row; shifted inputs staged (branch-free) to LDS
// transposed to [pixel][channel]; B fragments cached in VGPRs across all
// 20 oc tiles; A streamed from L2-resident bf16 W_eff; NT output stores.

typedef __bf16 bf16_t;
typedef __attribute__((ext_vector_type(4)))  __bf16 v4bf;
typedef __attribute__((ext_vector_type(8)))  __bf16 v8bf;
typedef __attribute__((ext_vector_type(16))) __bf16 v16bf;
typedef __attribute__((ext_vector_type(8)))  float  v8f;

#define C_CH   320          // channels (in = out)
#define IMG    128          // H = W
#define GSZ    64           // channels per shift group (320/5)
#define KTILES 10           // 320 / 32 (WMMA K)
#define OCT    20           // 320 / 16 (WMMA M tiles)
#define XS_STRIDE 328       // bf16 elems per LDS pixel row (320 + 8 pad; 656B, 16B-aligned)
#define SMEM_BYTES (IMG * XS_STRIDE * 2)   // 83,968 B  (< 320 KB/WGP)

union V16U { v16bf v; v8bf h[2]; };

// ---- prep: collapse masked 3x3 weights to dense 320x320 bf16 matrix --------
__global__ __launch_bounds__(256) void prep_weff(const float* __restrict__ w,
                                                 bf16_t* __restrict__ weff) {
  int idx = blockIdx.x * 256 + threadIdx.x;
  if (idx >= C_CH * C_CH) return;
  int oc = idx / C_CH;
  int ic = idx - oc * C_CH;
  int g  = ic / GSZ;                       // 0:right 1:left 2:down 3:up 4:center
  int ty = (g == 2) ? 2 : ((g == 3) ? 0 : 1);
  int tx = (g == 0) ? 2 : ((g == 1) ? 0 : 1);
  weff[idx] = (bf16_t)w[(((size_t)oc * C_CH + ic) * 3 + ty) * 3 + tx];
}

// ---- main: per-(n,h) row GEMM on WMMA ------------------------------------
__global__ __launch_bounds__(256) void shiftconv_wmma(
    const float* __restrict__ x, const bf16_t* __restrict__ weff,
    const float* __restrict__ bias, float* __restrict__ out) {
  extern __shared__ char smem_raw[];
  bf16_t* xs = (bf16_t*)smem_raw;          // [IMG pixels][XS_STRIDE channels] bf16

  const int h = blockIdx.x;                // 0..127
  const int n = blockIdx.y;                // 0..15
  const int t = threadIdx.x;               // 0..255 (8 waves)

  // ---- stage shifted input row into LDS, transposed to [pixel][channel] ---
  // Branch-free: clamp the source address (always-valid load) and multiply by
  // a per-group 0/1 mask -> no exec-mask churn in the hot loop.
  {
    const int wpx   = t & (IMG - 1);       // pixel column handled by this thread
    const int chalf = t >> 7;              // which channel quad parity
    const float* xn = x + (size_t)n * C_CH * IMG * IMG;
    for (int g = 0; g < 5; ++g) {
      const int dx = (g == 0) - (g == 1);
      const int dy = (g == 2) - (g == 3);
      const int sh = h + dy, sw = wpx + dx;
      const float msk = ((unsigned)sh < IMG && (unsigned)sw < IMG) ? 1.f : 0.f;
      const int shc = sh & (IMG - 1);      // wrap (-1 -> 127): valid addr, masked to 0
      const int swc = sw & (IMG - 1);
      const float* s = xn + ((size_t)(g * GSZ + chalf * 4) * IMG + shc) * IMG + swc;
      bf16_t*      d = xs + wpx * XS_STRIDE + g * GSZ + chalf * 4;
      #pragma unroll
      for (int q = 0; q < 8; ++q) {        // 8 quads of 4 channels, stride 8
        v4bf pv;
        #pragma unroll
        for (int j = 0; j < 4; ++j)
          pv[j] = (bf16_t)(s[(size_t)j * IMG * IMG] * msk);
        *(v4bf*)d = pv;                    // ds_store_b64
        s += (size_t)8 * IMG * IMG;
        d += 8;
      }
    }
  }
  __syncthreads();

  const int wave = t >> 5;                 // 8 waves <-> 8 pixel tiles of 16
  const int lane = t & 31;
  const int m    = lane & 15;              // row (A) / column-pixel (B,C)
  const int half = lane >> 4;              // K-half selector per ISA layout
  const int px0  = wave * 16;

  // ---- B fragments for the full K=320, cached in VGPRs (80 regs) ----------
  // B layout used here (contiguous K, same as A): lane n = lane%16; lanes 0-15 hold K=kc..kc+15,
  // lanes 16-31 hold K=kc+16..kc+31; VGPR v holds pair (2v,2v+1).
  v16bf bfrag[KTILES];
  {
    const bf16_t* brow = xs + (px0 + m) * XS_STRIDE + half * 16;
    #pragma unroll
    for (int kk = 0; kk < KTILES; ++kk) {
      V16U b;
      b.h[0] = *(const v8bf*)(brow + kk * 32);        // ds_read_b128
      b.h[1] = *(const v8bf*)(brow + kk * 32 + 8);    // ds_read_b128
      bfrag[kk] = b.v;
    }
  }

  // ---- 20 oc tiles x 10 K-steps of v_wmma_f32_16x16x32_bf16 ---------------
  const size_t outbase = (((size_t)n * C_CH) * IMG + h) * IMG;
  for (int ot = 0; ot < OCT; ++ot) {
    const int oc0 = ot * 16;
    v8f acc0 = {}, acc1 = {};              // two chains: break WMMA RAW serialization
    // A layout (matches B): lane m = lane%16; VGPRs 0-7 hold K = 16*half..+15.
    // W_eff is L2-resident (205 KB).
    const bf16_t* arow = weff + (size_t)(oc0 + m) * C_CH + 16 * half;
    #pragma unroll
    for (int kk = 0; kk < KTILES; kk += 2) {
      V16U a0, a1;
      a0.h[0] = *(const v8bf*)(arow + kk * 32);            // global_load b128
      a0.h[1] = *(const v8bf*)(arow + kk * 32 + 8);
      a1.h[0] = *(const v8bf*)(arow + kk * 32 + 32);
      a1.h[1] = *(const v8bf*)(arow + kk * 32 + 40);
      acc0 = __builtin_amdgcn_wmma_f32_16x16x32_bf16(
          false, a0.v, false, bfrag[kk],     (short)0, acc0, false, false);
      acc1 = __builtin_amdgcn_wmma_f32_16x16x32_bf16(
          false, a1.v, false, bfrag[kk + 1], (short)0, acc1, false, false);
    }
    const v8f acc = acc0 + acc1;
    // ---- bias + NT store: C VGPR r holds (M = r + 8*half, N = m) ----------
    const int pw = px0 + m;
    #pragma unroll
    for (int r = 0; r < 8; ++r) {
      const int oc = oc0 + r + 8 * half;
      __builtin_nontemporal_store(acc[r] + bias[oc],
          out + outbase + (size_t)oc * IMG * IMG + pw);
    }
  }
}

extern "C" void kernel_launch(void* const* d_in, const int* in_sizes, int n_in,
                              void* d_out, int out_size, void* d_ws, size_t ws_size,
                              hipStream_t stream) {
  (void)in_sizes; (void)n_in; (void)out_size; (void)ws_size;
  const float* x = (const float*)d_in[0];   // (16, 320, 128, 128) f32
  const float* w = (const float*)d_in[1];   // (320, 320, 3, 3)    f32
  const float* b = (const float*)d_in[2];   // (320,)              f32
  float* out     = (float*)d_out;           // (16, 320, 128, 128) f32
  bf16_t* weff   = (bf16_t*)d_ws;           // 320*320*2 = 204,800 B scratch

  prep_weff<<<(C_CH * C_CH + 255) / 256, 256, 0, stream>>>(w, weff);

  dim3 grid(IMG, 16);                       // (h, n): 2048 blocks
  shiftconv_wmma<<<grid, 256, SMEM_BYTES, stream>>>(x, weff, b, out);
}
